// CosmicNeuralNetwork_50964081935174
// MI455X (gfx1250) — matrix operand order repaired and checked by reading, not compiled, not run on hardware
//
#include <hip/hip_runtime.h>
#include <hip/hip_bf16.h>

typedef __attribute__((ext_vector_type(16))) __bf16 v16bf;
typedef __attribute__((ext_vector_type(8)))  float  v8f;

#define TILE_M 128
#define TILE_N 128
#define TILE_K 32

union FragU { v16bf v; uint4 q[2]; };

// ---------------- fp32 -> bf16 elementwise ----------------
__global__ __launch_bounds__(256) void convert_f32_to_bf16(
    const float* __restrict__ src, __bf16* __restrict__ dst, int n) {
  int i = blockIdx.x * blockDim.x + threadIdx.x;
  if (i < n) dst[i] = (__bf16)src[i];
}

// ---------------- W[K][N] fp32 -> Wt[N][K] bf16 (LDS tiled transpose) ----
__global__ __launch_bounds__(256) void transpose_convert(
    const float* __restrict__ W, __bf16* __restrict__ Wt, int K, int N) {
  __shared__ float tile[32][33];
  int tx = threadIdx.x & 31;
  int ty = threadIdx.x >> 5;               // 0..7
  int n0 = blockIdx.x * 32;
  int k0 = blockIdx.y * 32;
#pragma unroll
  for (int i = 0; i < 4; ++i)
    tile[ty + i * 8][tx] = W[(size_t)(k0 + ty + i * 8) * N + n0 + tx];
  __syncthreads();
#pragma unroll
  for (int i = 0; i < 4; ++i)
    Wt[(size_t)(n0 + ty + i * 8) * K + k0 + tx] = (__bf16)tile[tx][ty + i * 8];
}

// ---------------- cosmic: banded circulant stencil, one thread per row ----
// Only column 0 of the 2048-wide state survives; 24 total +/-1 shifts means a
// 51-wide window (center=25) is exact (s0 is constant across columns).
__global__ __launch_bounds__(256) void cosmic_kernel(
    const float* __restrict__ cw, float* __restrict__ cosv, int H) {
  int r = blockIdx.x * blockDim.x + threadIdx.x;
  if (r >= H) return;
  const float inv = 1.0f / sqrtf((float)H);
  float s[51], t[51];
#pragma unroll
  for (int j = 0; j < 51; ++j) s[j] = inv;
  for (int d = 0; d < 8; ++d) {
#pragma unroll
    for (int k = 0; k < 3; ++k) {
      float a  = cw[((size_t)d * H + r) * H + k];
      float c  = cosf(a);
      float si = sinf(a);
      t[0] = c * s[0] - si * s[1] + si * s[0];
#pragma unroll
      for (int j = 1; j < 50; ++j)
        t[j] = c * s[j] - si * s[j + 1] + si * s[j - 1];
      t[50] = c * s[50] - si * s[50] + si * s[49];
#pragma unroll
      for (int j = 0; j < 51; ++j) s[j] = t[j];
    }
  }
  cosv[r] = s[25] * s[25];
}

// ---------------- WMMA bf16 GEMM, 128x128 tile, BK=32, 8 waves -----------
// EPI: 0 = relu -> bf16, 1 = tanh(acc + bias + cos) -> bf16, 2 = relu -> f32
template <int EPI>
__global__ __launch_bounds__(256) void gemm_bf16_wmma(
    const __bf16* __restrict__ A,    // [M][K] row-major
    const __bf16* __restrict__ Wt,   // [N][K] row-major (pre-transposed)
    const float* __restrict__ bias,  // [N]
    const float* __restrict__ cosv,  // [N] (EPI==1 only)
    void* __restrict__ outp, int M, int N, int K) {
  __shared__ __bf16 As[TILE_M][TILE_K];
  __shared__ __bf16 Bs[TILE_N][TILE_K];

  const int t    = threadIdx.x;
  const int lane = t & 31;
  const int wave = t >> 5;
  const int wm   = wave & 3;   // M offset wm*32
  const int wn   = wave >> 2;  // N offset wn*64
  const int hlf  = lane >> 4;
  const int l15  = lane & 15;

  const int m0 = blockIdx.y * TILE_M;
  const int n0 = blockIdx.x * TILE_N;

  const int lrow = t >> 1;   // 0..127
  const int seg  = t & 1;    // 16 bf16 per segment

  const __bf16* gA = A  + (size_t)(m0 + lrow) * K + seg * 16;
  const __bf16* gB = Wt + (size_t)(n0 + lrow) * K + seg * 16;

  v8f acc[2][4] = {};

  for (int k0 = 0; k0 < K; k0 += TILE_K) {
    // global -> LDS (coalesced 16B loads)
    const uint4* sa = (const uint4*)(gA + k0);
    const uint4* sb = (const uint4*)(gB + k0);
    uint4 a0 = sa[0], a1 = sa[1];
    uint4 b0 = sb[0], b1 = sb[1];
    uint4* da = (uint4*)&As[lrow][seg * 16];
    da[0] = a0; da[1] = a1;
    uint4* db = (uint4*)&Bs[lrow][seg * 16];
    db[0] = b0; db[1] = b1;
    if (k0 + TILE_K < K) {
      __builtin_prefetch(gA + k0 + TILE_K, 0, 1);
      __builtin_prefetch(gB + k0 + TILE_K, 0, 1);
    }
    __syncthreads();

    // fragments per ISA 16-bit layouts:
    // A: lane<16 -> K[0..7]+K[16..23], lane>=16 -> K[8..15]+K[24..31]
    // B: lane<16 -> K[0..7]+K[16..23], lane>=16 -> K[8..15]+K[24..31]
    FragU a[2], b[4];
#pragma unroll
    for (int m = 0; m < 2; ++m) {
      int row = wm * 32 + m * 16 + l15;
      a[m].q[0] = *(const uint4*)&As[row][hlf * 8];
      a[m].q[1] = *(const uint4*)&As[row][16 + hlf * 8];
    }
#pragma unroll
    for (int n = 0; n < 4; ++n) {
      int col = wn * 64 + n * 16 + l15;
      b[n].q[0] = *(const uint4*)&Bs[col][hlf * 8];
      b[n].q[1] = *(const uint4*)&Bs[col][16 + hlf * 8];
    }
#pragma unroll
    for (int m = 0; m < 2; ++m)
#pragma unroll
      for (int n = 0; n < 4; ++n)
        acc[m][n] = __builtin_amdgcn_wmma_f32_16x16x32_bf16(
            false, a[m].v, false, b[n].v, (short)0, acc[m][n], false, false);
    __syncthreads();
  }

  // epilogue: C/D layout -> lane holds col (l15), rows hlf*8 + i
#pragma unroll
  for (int n = 0; n < 4; ++n) {
    int gn = n0 + wn * 64 + n * 16 + l15;
    float bv = bias[gn];
    float cv = (EPI == 1) ? cosv[gn] : 0.0f;
#pragma unroll
    for (int m = 0; m < 2; ++m) {
      int gmb = m0 + wm * 32 + m * 16 + hlf * 8;
#pragma unroll
      for (int i = 0; i < 8; ++i) {
        float f = acc[m][n][i] + bv;
        if (EPI == 1) f = tanhf(f + cv);
        else          f = fmaxf(f, 0.0f);
        if (EPI == 2) ((float*)outp)[(size_t)(gmb + i) * N + gn] = f;
        else          ((__bf16*)outp)[(size_t)(gmb + i) * N + gn] = (__bf16)f;
      }
    }
  }
}

extern "C" void kernel_launch(void* const* d_in, const int* in_sizes, int n_in,
                              void* d_out, int out_size, void* d_ws, size_t ws_size,
                              hipStream_t stream) {
  const int Bsz = 4096, IN = 1024, H = 2048, OUT = 1024;
  const float* x   = (const float*)d_in[0];
  const float* W0  = (const float*)d_in[1];
  const float* b0  = (const float*)d_in[2];
  const float* W1  = (const float*)d_in[3];
  const float* b1  = (const float*)d_in[4];
  const float* cw1 = (const float*)d_in[5];
  const float* KW1 = (const float*)d_in[7];
  const float* Kb1 = (const float*)d_in[8];
  const float* W2  = (const float*)d_in[9];
  const float* b2  = (const float*)d_in[10];
  const float* cw2 = (const float*)d_in[11];
  const float* KW2 = (const float*)d_in[13];
  const float* Kb2 = (const float*)d_in[14];
  const float* W3  = (const float*)d_in[15];
  const float* b3  = (const float*)d_in[16];

  char* ws = (char*)d_ws;
  size_t off = 0;
  auto alloc = [&](size_t bytes) {
    void* p = ws + off;
    off += (bytes + 255) & ~(size_t)255;
    return p;
  };
  __bf16* xb   = (__bf16*)alloc((size_t)Bsz * IN * 2);
  __bf16* Wt0  = (__bf16*)alloc((size_t)H * IN * 2);
  __bf16* Wt1  = (__bf16*)alloc((size_t)H * H * 2);
  __bf16* KWt1 = (__bf16*)alloc((size_t)H * H * 2);
  __bf16* Wt2  = (__bf16*)alloc((size_t)H * H * 2);
  __bf16* KWt2 = (__bf16*)alloc((size_t)H * H * 2);
  __bf16* Wt3  = (__bf16*)alloc((size_t)OUT * H * 2);
  __bf16* hA   = (__bf16*)alloc((size_t)Bsz * H * 2);
  __bf16* hB   = (__bf16*)alloc((size_t)Bsz * H * 2);
  float*  cos1 = (float*)alloc((size_t)H * 4);
  float*  cos2 = (float*)alloc((size_t)H * 4);

  int nx = Bsz * IN;
  convert_f32_to_bf16<<<(nx + 255) / 256, 256, 0, stream>>>(x, xb, nx);

  transpose_convert<<<dim3(H / 32, IN / 32), 256, 0, stream>>>(W0, Wt0, IN, H);
  transpose_convert<<<dim3(H / 32, H / 32), 256, 0, stream>>>(W1, Wt1, H, H);
  transpose_convert<<<dim3(H / 32, H / 32), 256, 0, stream>>>(KW1, KWt1, H, H);
  transpose_convert<<<dim3(H / 32, H / 32), 256, 0, stream>>>(W2, Wt2, H, H);
  transpose_convert<<<dim3(H / 32, H / 32), 256, 0, stream>>>(KW2, KWt2, H, H);
  transpose_convert<<<dim3(OUT / 32, H / 32), 256, 0, stream>>>(W3, Wt3, H, OUT);

  cosmic_kernel<<<H / 256, 256, 0, stream>>>(cw1, cos1, H);
  cosmic_kernel<<<H / 256, 256, 0, stream>>>(cw2, cos2, H);

  // h1 = relu(x@W0+b0)
  gemm_bf16_wmma<0><<<dim3(H / 128, Bsz / 128), 256, 0, stream>>>(xb, Wt0, b0, nullptr, hA, Bsz, H, IN);
  // h2 = relu(h1@W1+b1)
  gemm_bf16_wmma<0><<<dim3(H / 128, Bsz / 128), 256, 0, stream>>>(hA, Wt1, b1, nullptr, hB, Bsz, H, H);
  // h3 = tanh(h2@KW1+Kb1+cos1)
  gemm_bf16_wmma<1><<<dim3(H / 128, Bsz / 128), 256, 0, stream>>>(hB, KWt1, Kb1, cos1, hA, Bsz, H, H);
  // h4 = relu(h3@W2+b2)
  gemm_bf16_wmma<0><<<dim3(H / 128, Bsz / 128), 256, 0, stream>>>(hA, Wt2, b2, nullptr, hB, Bsz, H, H);
  // h5 = tanh(h4@KW2+Kb2+cos2)
  gemm_bf16_wmma<1><<<dim3(H / 128, Bsz / 128), 256, 0, stream>>>(hB, KWt2, Kb2, cos2, hA, Bsz, H, H);
  // out = relu(h5@W3+b3) -> fp32
  gemm_bf16_wmma<2><<<dim3(OUT / 128, Bsz / 128), 256, 0, stream>>>(hA, Wt3, b3, nullptr, d_out, Bsz, OUT, H);
}
